// Attention_50989851738495
// MI455X (gfx1250) — matrix operand order repaired and checked by reading, not compiled, not run on hardware
//
#include <hip/hip_runtime.h>
#include <hip/hip_bf16.h>

#define B_  2
#define S_  2048
#define D_  2048
#define H_  16
#define DH_ 128

typedef _Float16 h8   __attribute__((ext_vector_type(8)));
typedef _Float16 h16  __attribute__((ext_vector_type(16)));
typedef float    f8   __attribute__((ext_vector_type(8)));

#define WMMA_F16(a,b,c) \
  __builtin_amdgcn_wmma_f32_16x16x32_f16(false,(a),false,(b),(short)0,(c),false,false)

__device__ __forceinline__ f8 f8zero() {
  f8 z;
#pragma unroll
  for (int i = 0; i < 8; ++i) z[i] = 0.0f;
  return z;
}

// 16 contiguous halves -> h16 (two b128 loads)
__device__ __forceinline__ h16 ld16(const _Float16* p) {
  h8 lo = *(const h8*)(p);
  h8 hi = *(const h8*)(p + 8);
  h16 r;
#pragma unroll
  for (int i = 0; i < 8; ++i) { r[i] = lo[i]; r[i + 8] = hi[i]; }
  return r;
}

// two disjoint runs of 8 halves -> h16
__device__ __forceinline__ h16 ld8x2(const _Float16* p0, const _Float16* p1) {
  h8 a = *(const h8*)(p0);
  h8 b = *(const h8*)(p1);
  h16 r;
#pragma unroll
  for (int i = 0; i < 8; ++i) { r[i] = a[i]; r[i + 8] = b[i]; }
  return r;
}

// ---------------------------------------------------------------- convert f32 -> f16
__global__ __launch_bounds__(256) void cvt_h_kernel(const float* __restrict__ src,
                                                    _Float16* __restrict__ dst, int n) {
  int i = blockIdx.x * 256 + threadIdx.x;
  if (i < n) dst[i] = (_Float16)src[i];
}

// ---------------------------------------------------------------- C(f32) = A(f16,MxK) @ W(f16,NxK)^T
// block: 256 thr = 8 waves, macro tile 128(M) x 128(N); wave tile 32x64 (2x4 frags)
__global__ __launch_bounds__(256) void gemm_awt_kernel(const _Float16* __restrict__ A,
                                                       const _Float16* __restrict__ W,
                                                       float* __restrict__ C,
                                                       int M, int N, int K) {
  const int lane = threadIdx.x & 31;
  const int wid  = threadIdx.x >> 5;
  const int wm   = wid & 3;
  const int wn   = wid >> 2;
  const int row0 = blockIdx.y * 128 + wm * 32;
  const int col0 = blockIdx.x * 128 + wn * 64;
  const int l16  = lane & 15;
  const bool hi  = lane >= 16;

  f8 acc[2][4];
#pragma unroll
  for (int mf = 0; mf < 2; ++mf)
#pragma unroll
    for (int nf = 0; nf < 4; ++nf) acc[mf][nf] = f8zero();

  for (int kk = 0; kk < K; kk += 32) {
    h16 a[2];
#pragma unroll
    for (int mf = 0; mf < 2; ++mf) {
      // A layout: M = lane&15 ; lanes<16: K = kk..kk+7, kk+16..kk+23 ; lanes>=16: +8
      const _Float16* ap = A + (size_t)(row0 + mf * 16 + l16) * K + kk + (hi ? 8 : 0);
      a[mf] = ld8x2(ap, ap + 16);
    }
#pragma unroll
    for (int nf = 0; nf < 4; ++nf) {
      // B layout: N = lane&15 ; lanes<16: K = kk..kk+7, kk+16..kk+23 ; lanes>=16: +8
      const _Float16* bp = W + (size_t)(col0 + nf * 16 + l16) * K + kk + (hi ? 8 : 0);
      h16 b = ld8x2(bp, bp + 16);
      acc[0][nf] = WMMA_F16(a[0], b, acc[0][nf]);
      acc[1][nf] = WMMA_F16(a[1], b, acc[1][nf]);
    }
  }

  // C layout: lanes<16 -> (M=r, N=l16) ; lanes>=16 -> (M=r+8, N=l16)
#pragma unroll
  for (int mf = 0; mf < 2; ++mf)
#pragma unroll
    for (int nf = 0; nf < 4; ++nf)
#pragma unroll
      for (int r = 0; r < 8; ++r) {
        int m = row0 + mf * 16 + r + (hi ? 8 : 0);
        int n = col0 + nf * 16 + l16;
        C[(size_t)m * N + n] = acc[mf][nf][r];
      }
}

// ---------------------------------------------------------------- RoPE + [b,s,h,d] -> [b,h,s,d] (f16)
__global__ __launch_bounds__(256) void rope_pack_kernel(const float* __restrict__ G,
                                                        const float* __restrict__ cosT,
                                                        const float* __restrict__ sinT,
                                                        _Float16* __restrict__ out,
                                                        float scale) {
  int idx = blockIdx.x * 256 + threadIdx.x;      // B*S*H*(DH/2) = 4194304
  int i = idx & 63;
  int h = (idx >> 6) & 15;
  int s = (idx >> 10) & 2047;
  int b = idx >> 21;
  size_t gi = (size_t)(b * S_ + s) * D_ + h * DH_ + 2 * i;
  float a  = G[gi];
  float bb = G[gi + 1];
  float c  = cosT[s * 64 + i];
  float sn = sinT[s * 64 + i];
  size_t o = ((size_t)(b * H_ + h) * S_ + s) * DH_ + 2 * i;
  out[o]     = (_Float16)((a * c - bb * sn) * scale);
  out[o + 1] = (_Float16)((a * sn + bb * c) * scale);
}

// ---------------------------------------------------------------- V: [b,s,h,d] f32 -> [b,h,d,s] f16
__global__ __launch_bounds__(256) void v_pack_kernel(const float* __restrict__ G,
                                                     _Float16* __restrict__ Vt) {
  int idx = blockIdx.x * 256 + threadIdx.x;      // B*H*DH*S = 8388608
  int s = idx & 2047;
  int d = (idx >> 11) & 127;
  int h = (idx >> 18) & 15;
  int b = idx >> 22;
  Vt[idx] = (_Float16)G[(size_t)(b * S_ + s) * D_ + h * DH_ + d];
}

// ---------------------------------------------------------------- flash attention
// one wave = 16 queries of one (b,h). S^T = K.Q^T so softmax rows live in lanes.
__global__ __launch_bounds__(256) void flash_kernel(const _Float16* __restrict__ Qh,
                                                    const _Float16* __restrict__ Kh,
                                                    const _Float16* __restrict__ Vt,
                                                    _Float16* __restrict__ Ctx) {
  __shared__ _Float16 Plds[8 * 16 * 32];         // 8 KB: per-wave 16x32 P tile
  const int lane = threadIdx.x & 31;
  const int wid  = threadIdx.x >> 5;
  const int gw   = blockIdx.x * 8 + wid;         // 0..4095
  const int bh   = gw >> 7;                      // b*H + h
  const int q0   = (gw & 127) * 16;
  const int b    = bh >> 4;
  const int h    = bh & 15;
  const int l16  = lane & 15;
  const bool hi  = lane >= 16;
  _Float16* Pw = &Plds[wid * 16 * 32];

  const _Float16* Qb = Qh + (size_t)bh * S_ * DH_;
  const _Float16* Kb = Kh + (size_t)bh * S_ * DH_;
  const _Float16* Vb = Vt + (size_t)bh * DH_ * S_;

  // Q as B-fragments of S^T = K.Q^T : N = query = lane&15, K = d
  h16 qfrag[4];
#pragma unroll
  for (int dk = 0; dk < 4; ++dk)
    qfrag[dk] = ld8x2(Qb + (size_t)(q0 + l16) * DH_ + dk * 32 + (hi ? 8 : 0), Qb + (size_t)(q0 + l16) * DH_ + dk * 32 + (hi ? 8 : 0) + 16);

  f8 o[8];
#pragma unroll
  for (int dt = 0; dt < 8; ++dt) o[dt] = f8zero();
  float m = -1e30f, l = 0.0f;

  const int nkb = (q0 + 15) / 32 + 1;            // causal: keys 0..q0+15
  for (int kb = 0; kb < nkb; ++kb) {
    const int key0 = kb * 32;

    // two 16-key score tiles: st[t] C-layout (M=key_local, N=query)
    f8 st[2];
#pragma unroll
    for (int t = 0; t < 2; ++t) {
      st[t] = f8zero();
      const int kt = key0 + t * 16;
#pragma unroll
      for (int dk = 0; dk < 4; ++dk) {
        const _Float16* kp = Kb + (size_t)(kt + l16) * DH_ + dk * 32 + (hi ? 8 : 0);
        h16 ka = ld8x2(kp, kp + 16);
        st[t] = WMMA_F16(ka, qfrag[dk], st[t]);
      }
    }

    // causal mask + running max (row = query = l16; 16 of 32 keys in this lane)
    const int q = q0 + l16;
    float tmax = -1e30f;
#pragma unroll
    for (int t = 0; t < 2; ++t)
#pragma unroll
      for (int r = 0; r < 8; ++r) {
        int key = key0 + t * 16 + r + (hi ? 8 : 0);
        float v = (key <= q) ? st[t][r] : -1e30f;
        st[t][r] = v;
        tmax = fmaxf(tmax, v);
      }
    tmax = fmaxf(tmax, __shfl_xor(tmax, 16, 32));
    float mnew  = fmaxf(m, tmax);
    float alpha = __expf(m - mnew);

    float psum = 0.0f;
    h8 pv[2];
#pragma unroll
    for (int t = 0; t < 2; ++t)
#pragma unroll
      for (int r = 0; r < 8; ++r) {
        float p = __expf(st[t][r] - mnew);
        psum += p;
        pv[t][r] = (_Float16)p;
      }
    psum += __shfl_xor(psum, 16, 32);
    l = l * alpha + psum;
    m = mnew;

    // P (16q x 32k) to LDS, row-major per wave (DS ops in-order within wave)
    *(h8*)(Pw + l16 * 32 +      (hi ? 8 : 0)) = pv[0];
    *(h8*)(Pw + l16 * 32 + 16 + (hi ? 8 : 0)) = pv[1];

    // per-output-row (M = r + 8*hi) rescale factors, gathered from stat lanes
    float ar[8];
#pragma unroll
    for (int r = 0; r < 8; ++r) ar[r] = __shfl(alpha, r + (hi ? 8 : 0), 32);

    // P back as f16 A-fragment: M = query = l16, K = key_local 0..31
    const _Float16* pp = Pw + l16 * 32 + (hi ? 8 : 0);
    h16 pa = ld8x2(pp, pp + 16);

#pragma unroll
    for (int dt = 0; dt < 8; ++dt) {
      f8 c;
#pragma unroll
      for (int r = 0; r < 8; ++r) c[r] = o[dt][r] * ar[r];
      // V^T B-fragment: N = d = dt*16 + l16, K = key in two runs of 8 along s (+8 for hi lanes)
      const _Float16* vp = Vb + (size_t)(dt * 16 + l16) * S_ + key0 + (hi ? 8 : 0);
      h16 vb = ld8x2(vp, vp + 16);
      o[dt] = WMMA_F16(pa, vb, c);
    }
  }

  // normalize by l (per output row) and store Ctx as [b,s,h,d] f16
  float li = 1.0f / l;
  float linv[8];
#pragma unroll
  for (int r = 0; r < 8; ++r) linv[r] = __shfl(li, r + (hi ? 8 : 0), 32);

#pragma unroll
  for (int dt = 0; dt < 8; ++dt)
#pragma unroll
    for (int r = 0; r < 8; ++r) {
      int srow = q0 + r + (hi ? 8 : 0);
      int d    = dt * 16 + l16;
      Ctx[((size_t)(b * S_ + srow) * H_ + h) * DH_ + d] = (_Float16)(o[dt][r] * linv[r]);
    }
}

// ---------------------------------------------------------------- host launcher
extern "C" void kernel_launch(void* const* d_in, const int* in_sizes, int n_in,
                              void* d_out, int out_size, void* d_ws, size_t ws_size,
                              hipStream_t stream) {
  const float* x    = (const float*)d_in[0];
  const float* fcos = (const float*)d_in[1];
  const float* fsin = (const float*)d_in[2];
  // d_in[3] = mask: causal, applied analytically
  const float* wq = (const float*)d_in[4];
  const float* wk = (const float*)d_in[5];
  const float* wv = (const float*)d_in[6];
  const float* wo = (const float*)d_in[7];
  float* out = (float*)d_out;

  const int BS = B_ * S_;                 // 4096
  const int NX = BS * D_;                 // 8388608
  const int NW = D_ * D_;                 // 4194304

  char* ws = (char*)d_ws;
  _Float16* xh  = (_Float16*)(ws);                              // 16 MB
  _Float16* wh  = (_Float16*)(ws + 16777216);                   //  8 MB (reused per weight)
  float*    gf  = (float*)   (ws + 25165824);                   // 32 MB (reused per projection)
  _Float16* qh  = (_Float16*)(ws + 58720256);                   // 16 MB
  _Float16* kh  = (_Float16*)(ws + 75497472);                   // 16 MB
  _Float16* vt  = (_Float16*)(ws + 92274688);                   // 16 MB
  _Float16* ctx = (_Float16*)(ws + 109051904);                  // 16 MB

  const dim3 gemm_grid(D_ / 128, BS / 128);   // (16, 32)
  const float scale = 0.08838834764831845f;   // 1/sqrt(128)

  cvt_h_kernel<<<NX / 256, 256, 0, stream>>>(x, xh, NX);

  // Q projection + RoPE (scale folded into Q)
  cvt_h_kernel<<<NW / 256, 256, 0, stream>>>(wq, wh, NW);
  gemm_awt_kernel<<<gemm_grid, 256, 0, stream>>>(xh, wh, gf, BS, D_, D_);
  rope_pack_kernel<<<(B_ * S_ * H_ * 64) / 256, 256, 0, stream>>>(gf, fcos, fsin, qh, scale);

  // K projection + RoPE
  cvt_h_kernel<<<NW / 256, 256, 0, stream>>>(wk, wh, NW);
  gemm_awt_kernel<<<gemm_grid, 256, 0, stream>>>(xh, wh, gf, BS, D_, D_);
  rope_pack_kernel<<<(B_ * S_ * H_ * 64) / 256, 256, 0, stream>>>(gf, fcos, fsin, kh, 1.0f);

  // V projection + transpose to [b,h,d,s]
  cvt_h_kernel<<<NW / 256, 256, 0, stream>>>(wv, wh, NW);
  gemm_awt_kernel<<<gemm_grid, 256, 0, stream>>>(xh, wh, gf, BS, D_, D_);
  v_pack_kernel<<<(B_ * H_ * DH_ * S_) / 256, 256, 0, stream>>>(gf, vt);

  // flash attention: 4096 waves, 8 per block
  flash_kernel<<<512, 256, 0, stream>>>(qh, kh, vt, ctx);

  // output projection -> f32 d_out
  cvt_h_kernel<<<NW / 256, 256, 0, stream>>>(wo, wh, NW);
  gemm_awt_kernel<<<gemm_grid, 256, 0, stream>>>(ctx, wh, out, BS, D_, D_);
}
